// CPG_ConvE_10316511445861
// MI455X (gfx1250) — matrix operand order repaired and checked by reading, not compiled, not run on hardware
//
#include <hip/hip_runtime.h>
#include <math.h>

// ---------------- constants ----------------
constexpr int BSZ   = 256;
constexpr int EDIM  = 128;
constexpr int HIDN  = 256;
constexpr int FEAT  = 2688;          // 32 * 6 * 14
constexpr int NENT  = 40943;
constexpr long long W2COLS = 344064; // FEAT * EDIM
constexpr float EPSV = 1e-5f;

// ---------------- WMMA types ----------------
typedef __attribute__((ext_vector_type(16))) __bf16 v16bf;
typedef __attribute__((ext_vector_type(8)))  float  v8f;
typedef __attribute__((ext_vector_type(4)))  float  vf4;   // native vector (builtin-friendly)

union ABTile { uint4 q[2]; unsigned short s[16]; v16bf v; };

// Native bf16 convert (lowers to v_cvt_pk_bf16_f32 on gfx1250).
__device__ inline unsigned short f2bf(float f) {
  __bf16 h = (__bf16)f;
  return __builtin_bit_cast(unsigned short, h);
}

__device__ inline v8f wmma_bf16(const ABTile& a, const ABTile& b, v8f c) {
  return __builtin_amdgcn_wmma_f32_16x16x32_bf16(false, a.v, false, b.v,
                                                 (short)0, c, false, false);
}

// ---------------- gfx1250 async global->LDS copy (ASYNCcnt) ----------------
#if __has_builtin(__builtin_amdgcn_global_load_async_to_lds_b128)
#define HAVE_ASYNC_LDS 1
typedef int v4i __attribute__((vector_size(16)));
typedef __attribute__((address_space(1))) v4i* v4i_gp;   // global int4*
typedef __attribute__((address_space(3))) v4i* v4i_lp;   // LDS int4*
__device__ inline void async_cp16(void* g, void* l) {
  __builtin_amdgcn_global_load_async_to_lds_b128((v4i_gp)g, (v4i_lp)l, 0, 0);
}
__device__ inline void async_wait0() {
#if __has_builtin(__builtin_amdgcn_s_wait_asynccnt)
  __builtin_amdgcn_s_wait_asynccnt(0);
#else
  asm volatile("s_wait_asynccnt 0x0" ::: "memory");
#endif
}
#else
#define HAVE_ASYNC_LDS 0
#endif

// ---------------- gfx1250 Tensor Data Mover (TENSORcnt) ----------------
#if __has_builtin(__builtin_amdgcn_tensor_load_to_lds) && \
    __has_builtin(__builtin_amdgcn_s_wait_tensorcnt)
#define HAVE_TDM 1
typedef unsigned u32x4 __attribute__((vector_size(16)));
typedef int      i32x4 __attribute__((vector_size(16)));
typedef int      i32x8 __attribute__((vector_size(32)));
typedef __attribute__((address_space(3))) void* as3_void;

// 2D tile load: 16 dwords/row x 256 rows, row stride 1344 dwords,
// LDS pad 4 dwords after every 16 dwords (-> 80-byte LDS rows).
__device__ inline void tdm_load_x(unsigned lds_off, unsigned long long ga) {
  u32x4 g0 = { 1u,                      // count=1 valid descriptor
               lds_off,                 // lds_addr (bytes)
               (unsigned)ga,            // global_addr[31:0]
               (unsigned)(ga >> 32) | (2u << 30) };  // addr[56:32] | type=2
  i32x8 g1 = {
    (int)((2u << 16) | (1u << 20) | (3u << 22) | (3u << 25)),
    //     data_size=4B | pad_enable | pad_interval=16dw | pad_amount=4dw
    (int)(16u << 16),                  // tensor_dim0 = 16 (bits 63:48)
    (int)(256u << 16),                 // tensor_dim1 = 256 (bits 95:80)
    (int)(16u << 16),                  // tile_dim0 = 16 (bits 127:112)
    (int)256,                          // tile_dim1 = 256, tile_dim2 = 0
    (int)1344,                         // tensor_dim0_stride = 1344 dwords
    0, 0 };
  i32x4 z4 = {0, 0, 0, 0};
#if __clang_major__ >= 23
  i32x8 z8 = {0, 0, 0, 0, 0, 0, 0, 0};
  __builtin_amdgcn_tensor_load_to_lds(g0, g1, z4, z4, z8, 0);
#else
  __builtin_amdgcn_tensor_load_to_lds(g0, g1, z4, z4, 0);
#endif
}
#else
#define HAVE_TDM 0
#endif

// ============================================================
// K1: gather R rows, gather E1 rows, bn0 (global mean/var) on E1
// ============================================================
__global__ __launch_bounds__(1024)
void k1_gather_bn0(const int* __restrict__ e1, const int* __restrict__ r,
                   const float* __restrict__ ent, const float* __restrict__ rel,
                   float* __restrict__ Rg, float* __restrict__ E1n) {
  __shared__ float ssum[1024], ssq[1024];
  __shared__ float mI[2];
  int t = threadIdx.x;
  float s = 0.f, q = 0.f;
  for (int i = t; i < BSZ * EDIM; i += 1024) {
    int bi = i >> 7, k = i & 127;
    Rg[i] = rel[r[bi] * EDIM + k];
    float v = ent[(long long)e1[bi] * EDIM + k];
    E1n[i] = v;
    s += v; q += v * v;
  }
  ssum[t] = s; ssq[t] = q; __syncthreads();
  for (int off = 512; off > 0; off >>= 1) {
    if (t < off) { ssum[t] += ssum[t + off]; ssq[t] += ssq[t + off]; }
    __syncthreads();
  }
  if (t == 0) {
    float m = ssum[0] / (float)(BSZ * EDIM);
    float var = ssq[0] / (float)(BSZ * EDIM) - m * m;
    mI[0] = m; mI[1] = rsqrtf(var + EPSV);
  }
  __syncthreads();
  float m = mI[0], inv = mI[1];
  for (int i = t; i < BSZ * EDIM; i += 1024) E1n[i] = (E1n[i] - m) * inv;
}

// ============================================================
// K2: CPG layer 1 (x4): H = relu(R @ w1), [256,128]x[128,256]
// ============================================================
__global__ __launch_bounds__(32)
void k2_cpg1(const float* __restrict__ Rg,
             const float* __restrict__ w1a, const float* __restrict__ w1b,
             const float* __restrict__ w1c, const float* __restrict__ w1d,
             float* __restrict__ Ha, float* __restrict__ Hb,
             float* __restrict__ Hc, float* __restrict__ Hd) {
  int lane = threadIdx.x;
  int nt = blockIdx.x, mt = blockIdx.y, z = blockIdx.z;
  const float* w1 = (z == 0) ? w1a : (z == 1) ? w1b : (z == 2) ? w1c : w1d;
  float* H = (z == 0) ? Ha : (z == 1) ? Hb : (z == 2) ? Hc : Hd;
  int m = mt * 16 + (lane & 15);
  int n = nt * 16 + (lane & 15);
  int ah = (lane & 16) ? 8 : 0;
  int bh = (lane & 16) ? 16 : 0;
  v8f acc = {0.f, 0.f, 0.f, 0.f, 0.f, 0.f, 0.f, 0.f};
  for (int ks = 0; ks < 4; ++ks) {
    ABTile a, b;
    const float* ap = Rg + m * EDIM + ks * 32;
#pragma unroll
    for (int i = 0; i < 8; ++i) { a.s[i] = f2bf(ap[bh + i]); a.s[8 + i] = f2bf(ap[bh + 8 + i]); }
#pragma unroll
    for (int i = 0; i < 16; ++i) b.s[i] = f2bf(w1[(ks * 32 + bh + i) * HIDN + n]);
    acc = wmma_bf16(a, b, acc);
  }
#pragma unroll
  for (int j = 0; j < 8; ++j) {
    int row = mt * 16 + j + ((lane & 16) ? 8 : 0);
    float v = acc[j]; v = v > 0.f ? v : 0.f;
    H[row * HIDN + n] = v;
  }
}

// ============================================================
// K3: CPG layer 2 small GEMMs (filt 288 / cbias 32 / fcb 128 cols)
// ============================================================
__global__ __launch_bounds__(32)
void k3_cpg2(const float* __restrict__ Hcw, const float* __restrict__ Hcb,
             const float* __restrict__ Hfb,
             const float* __restrict__ w2cw, const float* __restrict__ w2cb,
             const float* __restrict__ w2fb,
             float* __restrict__ filt, float* __restrict__ cbias,
             float* __restrict__ fcbo) {
  int lane = threadIdx.x;
  int id = blockIdx.x, mt = blockIdx.y;
  const float* Hm; const float* w2; float* outp; int N; int nt;
  if (id < 18)      { Hm = Hcw; w2 = w2cw; outp = filt;  N = 288; nt = id; }
  else if (id < 20) { Hm = Hcb; w2 = w2cb; outp = cbias; N = 32;  nt = id - 18; }
  else              { Hm = Hfb; w2 = w2fb; outp = fcbo;  N = 128; nt = id - 20; }
  int m = mt * 16 + (lane & 15);
  int n = nt * 16 + (lane & 15);
  int ah = (lane & 16) ? 8 : 0;
  int bh = (lane & 16) ? 16 : 0;
  v8f acc = {0.f, 0.f, 0.f, 0.f, 0.f, 0.f, 0.f, 0.f};
  for (int ks = 0; ks < 8; ++ks) {
    ABTile a, b;
    const float* ap = Hm + m * HIDN + ks * 32;
#pragma unroll
    for (int i = 0; i < 8; ++i) { a.s[i] = f2bf(ap[bh + i]); a.s[8 + i] = f2bf(ap[bh + 8 + i]); }
#pragma unroll
    for (int i = 0; i < 16; ++i) b.s[i] = f2bf(w2[(ks * 32 + bh + i) * N + n]);
    acc = wmma_bf16(a, b, acc);
  }
#pragma unroll
  for (int j = 0; j < 8; ++j) {
    int row = mt * 16 + j + ((lane & 16) ? 8 : 0);
    outp[row * N + n] = acc[j];
  }
}

// ============================================================
// K4: per-sample grouped conv 3x3 + bias + relu -> X bf16 [256,2688]
// LDS staging via GLOBAL_LOAD_ASYNC_TO_LDS_B128 (pure copies)
// ============================================================
__global__ __launch_bounds__(128)
void k4_conv(const float* __restrict__ E1n, const float* __restrict__ filt,
             const float* __restrict__ cbias, unsigned short* __restrict__ Xbf) {
  __shared__ __align__(16) float img[128];
  __shared__ __align__(16) float flt[288];
  __shared__ __align__(16) float cb[32];
  int b = blockIdx.x, t = threadIdx.x;
#if HAVE_ASYNC_LDS
  if (t < 32)       async_cp16((void*)(E1n + b * 128 + t * 4), &img[t * 4]);
  else if (t < 104) { int i = t - 32;  async_cp16((void*)(filt + b * 288 + i * 4), &flt[i * 4]); }
  else if (t < 112) { int i = t - 104; async_cp16((void*)(cbias + b * 32 + i * 4), &cb[i * 4]); }
  async_wait0();
#else
  img[t] = E1n[b * 128 + t];
  for (int i = t; i < 288; i += 128) flt[i] = filt[b * 288 + i];
  if (t < 32) cb[t] = cbias[b * 32 + t];
#endif
  __syncthreads();
  for (int o = t; o < FEAT; o += 128) {
    int c = o / 84, p = o % 84;
    int oy = p / 14, ox = p % 14;
    float s = cb[c];
#pragma unroll
    for (int ky = 0; ky < 3; ++ky)
#pragma unroll
      for (int kx = 0; kx < 3; ++kx)
        s += img[(oy + ky) * 16 + ox + kx] * flt[c * 9 + ky * 3 + kx];
    s = s > 0.f ? s : 0.f;
    Xbf[b * FEAT + o] = f2bf(s);
  }
}

// ============================================================
// K5: fused fc path (double-buffered software pipeline).
// For h = blockIdx.x:
//   T_h = X @ W_h  ([256,2688]x[2688,128], bf16 WMMA)
//   Y  += diag(Hfw[:,h]) @ T_h   (f32 global atomics)
// Per iteration: issue TDM tensor_load_to_lds for block fb+1 (X tile,
// D# pads each 64B row by 16B -> bank-spread LDS rows), NT-stream +
// convert W block fb+1, compute 8 WMMAs on block fb, then
// s_wait_tensorcnt + ONE barrier.  DMA latency hides under the XDL ops.
// ============================================================
__global__ __launch_bounds__(512)
void k5_fcw(const unsigned short* __restrict__ Xbf,
            const float* __restrict__ w2, const float* __restrict__ Hfw,
            float* __restrict__ Y) {
  __shared__ __align__(16) unsigned short Xs[2][256][40]; // 2 x 20 KB
  __shared__ __align__(16) unsigned short Ws[2][128][40]; // 2 x 10 KB
  int t = threadIdx.x, lane = t & 31, wave = t >> 5;
  int h = blockIdx.x;
  const float* w2h = w2 + (long long)h * W2COLS;

  float hs[8];
#pragma unroll
  for (int j = 0; j < 8; ++j)
    hs[j] = Hfw[(wave * 16 + j + ((lane & 16) ? 8 : 0)) * HIDN + h];

  v8f acc[8];
  v8f zv = {0.f, 0.f, 0.f, 0.f, 0.f, 0.f, 0.f, 0.f};
#pragma unroll
  for (int i = 0; i < 8; ++i) acc[i] = zv;

  int xrow = t >> 1, xhalf = t & 1;
  int arow = wave * 16 + (lane & 15);
  int ao = (lane & 16) ? 8 : 0;
  int bo = (lane & 16) ? 16 : 0;

#if HAVE_TDM
  unsigned xs_lds = (unsigned)(unsigned long long)(as3_void)(void*)&Xs[0][0][0];
  unsigned long long xg = (unsigned long long)(const void*)Xbf;
#endif

  // ---- staging helpers ----
  auto stage_x_issue = [&](int fb, int buf) {
#if HAVE_TDM
    if (wave == 0)
      tdm_load_x(xs_lds + (unsigned)buf * (unsigned)sizeof(Xs[0]),
                 xg + (unsigned long long)fb * 64ull);
#else
    unsigned short* gx =
        const_cast<unsigned short*>(Xbf) + xrow * FEAT + fb * 32 + xhalf * 16;
#if HAVE_ASYNC_LDS
    async_cp16(gx,     &Xs[buf][xrow][xhalf * 16]);
    async_cp16(gx + 8, &Xs[buf][xrow][xhalf * 16 + 8]);
#else
    uint4 q0 = ((const uint4*)gx)[0], q1 = ((const uint4*)gx)[1];
    *(uint4*)&Xs[buf][xrow][xhalf * 16]     = q0;
    *(uint4*)&Xs[buf][xrow][xhalf * 16 + 8] = q1;
#endif
#endif
  };
  auto stage_x_wait = [&]() {
#if HAVE_TDM
    if (wave == 0) __builtin_amdgcn_s_wait_tensorcnt(0);
#elif HAVE_ASYNC_LDS
    async_wait0();
#endif
  };
  auto stage_w = [&](int fb, int buf) {
    int flat = t * 8;
    int j = flat >> 7, c0 = flat & 127;
    const vf4* gw = (const vf4*)(w2h + fb * 4096 + flat);
    vf4 f0 = __builtin_nontemporal_load(gw);
    vf4 f1 = __builtin_nontemporal_load(gw + 1);
    Ws[buf][c0 + 0][j] = f2bf(f0[0]); Ws[buf][c0 + 1][j] = f2bf(f0[1]);
    Ws[buf][c0 + 2][j] = f2bf(f0[2]); Ws[buf][c0 + 3][j] = f2bf(f0[3]);
    Ws[buf][c0 + 4][j] = f2bf(f1[0]); Ws[buf][c0 + 5][j] = f2bf(f1[1]);
    Ws[buf][c0 + 6][j] = f2bf(f1[2]); Ws[buf][c0 + 7][j] = f2bf(f1[3]);
  };

  // ---- prologue: fill buffer 0 with block 0 ----
  stage_x_issue(0, 0);
  stage_w(0, 0);
  stage_x_wait();
  __syncthreads();

  for (int fb = 0; fb < 84; ++fb) {
    int cb = fb & 1, nb = cb ^ 1;
    if (fb + 1 < 84) {
      stage_x_issue(fb + 1, nb);          // TDM DMA runs under the WMMAs
      stage_w(fb + 1, nb);
      if (fb + 2 < 84)
        __builtin_prefetch(w2h + (fb + 2) * 4096 + t * 8, 0, 0);
    }
    // ---- compute on current buffer ----
    ABTile a;
    a.q[0] = *(const uint4*)&Xs[cb][arow][ao];
    a.q[1] = *(const uint4*)&Xs[cb][arow][ao + 16];
#pragma unroll
    for (int nt = 0; nt < 8; ++nt) {
      ABTile b;
      int c = nt * 16 + (lane & 15);
      b.q[0] = *(const uint4*)&Ws[cb][c][ao];
      b.q[1] = *(const uint4*)&Ws[cb][c][ao + 16];
      acc[nt] = wmma_bf16(a, b, acc[nt]);
    }
    if (fb + 1 < 84) stage_x_wait();
    __syncthreads();
  }

#pragma unroll
  for (int nt = 0; nt < 8; ++nt)
#pragma unroll
    for (int j = 0; j < 8; ++j) {
      int row = wave * 16 + j + ((lane & 16) ? 8 : 0);
      atomicAdd(&Y[row * EDIM + nt * 16 + (lane & 15)], acc[nt][j] * hs[j]);
    }
}

// ============================================================
// K6: Y += fcb bias; bn2 over batch per feature; relu -> X2 bf16
// ============================================================
__global__ __launch_bounds__(128)
void k6_bn2(const float* __restrict__ Y, const float* __restrict__ fcbo,
            unsigned short* __restrict__ X2bf) {
  int k = threadIdx.x;
  float s = 0.f, q = 0.f;
  for (int b = 0; b < BSZ; ++b) {
    float v = Y[b * EDIM + k] + fcbo[b * EDIM + k];
    s += v; q += v * v;
  }
  float m = s / (float)BSZ;
  float var = q / (float)BSZ - m * m;
  float inv = rsqrtf(var + EPSV);
  for (int b = 0; b < BSZ; ++b) {
    float v = (Y[b * EDIM + k] + fcbo[b * EDIM + k] - m) * inv;
    v = v > 0.f ? v : 0.f;
    X2bf[b * EDIM + k] = f2bf(v);
  }
}

// ============================================================
// K7: scoring GEMM  sigmoid(X2 @ E2^T + b)  [256,128]x[128,40943]
// ============================================================
__global__ __launch_bounds__(512)
void k7_score(const unsigned short* __restrict__ X2bf,
              const float* __restrict__ ent, const float* __restrict__ bias,
              float* __restrict__ out) {
  __shared__ __align__(16) unsigned short Es[16][136];   // padded (bank spread)
  __shared__ float bs[16];
  int t = threadIdx.x, lane = t & 31, wave = t >> 5;
  int e0 = blockIdx.x * 16;
  {
    int e = t >> 5, k = (t & 31) * 4;
    int eg = e0 + e; if (eg >= NENT) eg = NENT - 1;
    const vf4 f = *(const vf4*)(ent + (long long)eg * EDIM + k);
    unsigned p0 = (unsigned)f2bf(f[0]) | ((unsigned)f2bf(f[1]) << 16);
    unsigned p1 = (unsigned)f2bf(f[2]) | ((unsigned)f2bf(f[3]) << 16);
    *(unsigned*)&Es[e][k]     = p0;
    *(unsigned*)&Es[e][k + 2] = p1;
    if (t < 16) { int eb = e0 + t; bs[t] = (eb < NENT) ? bias[eb] : 0.f; }
  }
  __syncthreads();

  v8f acc = {0.f, 0.f, 0.f, 0.f, 0.f, 0.f, 0.f, 0.f};
  int m = wave * 16 + (lane & 15);
  int ao = (lane & 16) ? 8 : 0;
  int bo = (lane & 16) ? 16 : 0;
#pragma unroll
  for (int ks = 0; ks < 4; ++ks) {
    ABTile a, b;
    a.q[0] = *(const uint4*)(X2bf + m * EDIM + ks * 32 + ao);
    a.q[1] = *(const uint4*)(X2bf + m * EDIM + ks * 32 + ao + 16);
    b.q[0] = *(const uint4*)&Es[lane & 15][ks * 32 + ao];
    b.q[1] = *(const uint4*)&Es[lane & 15][ks * 32 + ao + 16];
    acc = wmma_bf16(a, b, acc);
  }
  int e = e0 + (lane & 15);
  if (e < NENT) {
#pragma unroll
    for (int j = 0; j < 8; ++j) {
      int row = wave * 16 + j + ((lane & 16) ? 8 : 0);
      float s = acc[j] + bs[lane & 15];
      __builtin_nontemporal_store(1.f / (1.f + __expf(-s)),
                                  &out[(long long)row * NENT + e]);
    }
  }
}

// ============================================================
// host launcher
// ============================================================
extern "C" void kernel_launch(void* const* d_in, const int* in_sizes, int n_in,
                              void* d_out, int out_size, void* d_ws, size_t ws_size,
                              hipStream_t stream) {
  const int*   e1       = (const int*)d_in[0];
  const int*   r        = (const int*)d_in[1];
  const float* ent      = (const float*)d_in[2];
  const float* rel      = (const float*)d_in[3];
  const float* convw_w1 = (const float*)d_in[4];
  const float* convw_w2 = (const float*)d_in[5];
  const float* convb_w1 = (const float*)d_in[6];
  const float* convb_w2 = (const float*)d_in[7];
  const float* fcw_w1   = (const float*)d_in[8];
  const float* fcw_w2   = (const float*)d_in[9];
  const float* fcb_w1   = (const float*)d_in[10];
  const float* fcb_w2   = (const float*)d_in[11];
  const float* bias     = (const float*)d_in[12];
  float* out = (float*)d_out;
  char* ws = (char*)d_ws;

  float* Rg    = (float*)(ws + 0);             // 256*128*4   = 131072
  float* E1n   = (float*)(ws + 131072);        // 131072
  float* Hcw   = (float*)(ws + 262144);        // 256*256*4   = 262144
  float* Hcb   = (float*)(ws + 524288);
  float* Hfw   = (float*)(ws + 786432);
  float* Hfb   = (float*)(ws + 1048576);
  float* filt  = (float*)(ws + 1310720);       // 256*288*4   = 294912
  float* cbias = (float*)(ws + 1605632);       // 256*32*4    = 32768
  float* fcbo  = (float*)(ws + 1638400);       // 131072
  unsigned short* Xbf  = (unsigned short*)(ws + 1769472);   // 256*2688*2 = 1376256
  float* Y     = (float*)(ws + 3145728);       // 131072
  unsigned short* X2bf = (unsigned short*)(ws + 3276800);   // 65536

  (void)hipMemsetAsync(Y, 0, BSZ * EDIM * sizeof(float), stream);

  k1_gather_bn0<<<1, 1024, 0, stream>>>(e1, r, ent, rel, Rg, E1n);
  k2_cpg1<<<dim3(16, 16, 4), 32, 0, stream>>>(Rg, convw_w1, convb_w1, fcw_w1,
                                              fcb_w1, Hcw, Hcb, Hfw, Hfb);
  k3_cpg2<<<dim3(28, 16), 32, 0, stream>>>(Hcw, Hcb, Hfb, convw_w2, convb_w2,
                                           fcb_w2, filt, cbias, fcbo);
  k4_conv<<<256, 128, 0, stream>>>(E1n, filt, cbias, Xbf);
  k5_fcw<<<256, 512, 0, stream>>>(Xbf, fcw_w2, Hfw, Y);
  k6_bn2<<<1, 128, 0, stream>>>(Y, fcbo, X2bf);
  k7_score<<<2559, 512, 0, stream>>>(X2bf, ent, bias, out);
}
